// SASRec_82918638617129
// MI455X (gfx1250) — matrix operand order repaired and checked by reading, not compiled, not run on hardware
//
#include <hip/hip_runtime.h>
#include <hip/hip_bf16.h>
#include <math.h>

// SASRec forward on gfx1250 (MI455X): bf16 WMMA GEMMs + flash attention.
#define B_   512
#define S_   200
#define SP_  224          // S padded to multiple of 32 for WMMA K-dim
#define D_   256
#define H_   4
#define DH_  64
#define F_   1024
#define L_   2
#define BS_  (B_ * S_)    // 102400 rows, multiple of 64

typedef __bf16 v16bf __attribute__((ext_vector_type(16)));
typedef float  v8f   __attribute__((ext_vector_type(8)));
typedef unsigned int u32x4 __attribute__((ext_vector_type(4)));  // 16B aligned

union FragU {
  v16bf v;
  unsigned short s[16];
  u32x4 q[2];
};

__device__ __forceinline__ unsigned short f2bf(float f) {
  union { float f; unsigned u; } c; c.f = f;
  unsigned r = c.u + 0x7FFFu + ((c.u >> 16) & 1u);   // round-to-nearest-even
  return (unsigned short)(r >> 16);
}

__device__ __forceinline__ v8f wmma_bf16(v16bf a, v16bf b, v8f c) {
  // (neg_a, A, neg_b, B, c_mod, C, reuse_a, reuse_b)
  return __builtin_amdgcn_wmma_f32_16x16x32_bf16(false, a, false, b, (short)0, c,
                                                 false, false);
}

__device__ __forceinline__ float warp_sum32(float v) {
#pragma unroll
  for (int m = 16; m >= 1; m >>= 1) v += __shfl_xor(v, m, 32);
  return v;
}
__device__ __forceinline__ float grp16_max(float v) {
#pragma unroll
  for (int m = 8; m >= 1; m >>= 1) v = fmaxf(v, __shfl_xor(v, m, 32));
  return v;
}
__device__ __forceinline__ float grp16_sum(float v) {
#pragma unroll
  for (int m = 8; m >= 1; m >>= 1) v += __shfl_xor(v, m, 32);
  return v;
}

// ---------------------------------------------------------------- convert
__global__ void f32_to_bf16_kernel(const float* __restrict__ src,
                                   unsigned short* __restrict__ dst, size_t n) {
  size_t i = (size_t)blockIdx.x * blockDim.x + threadIdx.x;
  size_t stride = (size_t)gridDim.x * blockDim.x;
  for (; i < n; i += stride) dst[i] = f2bf(src[i]);
}

// ---------------------------------------------------------------- layernorm
__device__ __forceinline__ void ln_write(float xr[8], const float* __restrict__ g,
                                         const float* __restrict__ bta,
                                         float* __restrict__ outf,
                                         unsigned short* __restrict__ outb,
                                         size_t ro, int lane) {
  float s = 0.f;
#pragma unroll
  for (int i = 0; i < 8; ++i) s += xr[i];
  float mean = warp_sum32(s) * (1.0f / 256.0f);
  float vs = 0.f;
#pragma unroll
  for (int i = 0; i < 8; ++i) { float d = xr[i] - mean; vs += d * d; }
  float var = warp_sum32(vs) * (1.0f / 256.0f);
  float rstd = rsqrtf(var + 1e-12f);
#pragma unroll
  for (int i = 0; i < 8; ++i) {
    int d = lane + i * 32;
    float y = (xr[i] - mean) * rstd * g[d] + bta[d];
    outf[ro + d] = y;
    outb[ro + d] = f2bf(y);
  }
}

__global__ __launch_bounds__(256) void ln_kernel(const float* __restrict__ in,
                                                 const float* __restrict__ g,
                                                 const float* __restrict__ bta,
                                                 float* __restrict__ outf,
                                                 unsigned short* __restrict__ outb) {
  int row = blockIdx.x * 8 + (threadIdx.x >> 5);
  int lane = threadIdx.x & 31;
  size_t ro = (size_t)row * D_;
  float xr[8];
#pragma unroll
  for (int i = 0; i < 8; ++i) xr[i] = in[ro + lane + i * 32];
  ln_write(xr, g, bta, outf, outb, ro, lane);
}

__global__ __launch_bounds__(256) void embed_ln_kernel(
    const float* __restrict__ item_emb, const float* __restrict__ pos_emb,
    const float* __restrict__ g, const float* __restrict__ bta,
    const int* __restrict__ seq, float* __restrict__ outf,
    unsigned short* __restrict__ outb) {
  int row = blockIdx.x * 8 + (threadIdx.x >> 5);   // row in [0, B*S)
  int lane = threadIdx.x & 31;
  int srow = row % S_;
  int item = seq[row];
  size_t ro = (size_t)row * D_;
  float xr[8];
#pragma unroll
  for (int i = 0; i < 8; ++i) {
    int d = lane + i * 32;
    xr[i] = item_emb[(size_t)item * D_ + d] + pos_emb[(size_t)srow * D_ + d];
  }
  ln_write(xr, g, bta, outf, outb, ro, lane);
}

// ---------------------------------------------------------------- GEMM
// C[M,N] = A[M,K](bf16) @ W[K,N](bf16) + bias, epilogue by FUSE:
//   0: -> bf16   1: exact GELU -> bf16   2: + resid -> f32
// Block tile 64x128, 4 waves, each wave 32x64 (2x4 WMMA tiles), K-step 32.
template <int FUSE>
__global__ __launch_bounds__(128) void gemm_bf16_kernel(
    const unsigned short* __restrict__ A, const unsigned short* __restrict__ W,
    const float* __restrict__ bias, const float* __restrict__ resid,
    unsigned short* __restrict__ outb, float* __restrict__ outf,
    int M, int N, int K) {
  __shared__ alignas(16) unsigned short As[64 * 32];     // [m][k] row-major
  __shared__ alignas(16) unsigned short BsT[128 * 32];   // [n][k] (transposed)
  const int tid = threadIdx.x;
  const int lane = tid & 31;
  const int wv = tid >> 5;              // 4 waves
  const int wm = (wv >> 1) << 5;        // wave row offset 0/32
  const int wn = (wv & 1) << 6;         // wave col offset 0/64
  const int hlf = lane >> 4;
  const int ln16 = lane & 15;
  const int bm = blockIdx.y << 6;
  const int bn = blockIdx.x << 7;
  (void)M;

  v8f acc[2][4];
#pragma unroll
  for (int i = 0; i < 2; ++i)
#pragma unroll
    for (int j = 0; j < 4; ++j)
#pragma unroll
      for (int v = 0; v < 8; ++v) acc[i][j][v] = 0.f;

  for (int k0 = 0; k0 < K; k0 += 32) {
    // stage A 64x32 row-major with b128 copies
#pragma unroll
    for (int it = 0; it < 2; ++it) {
      int j = tid + (it << 7);
      int r = j >> 1, part = j & 1;
      ((u32x4*)As)[j] =
          *(const u32x4*)(A + (size_t)(bm + r) * K + k0 + (part << 3));
    }
    // stage B 32x128 transposed into BsT[n][k] via register 2x2 transpose
#pragma unroll
    for (int it = 0; it < 8; ++it) {
      int j = tid + (it << 7);             // 1024 items: 16 k-pairs x 64 n-pairs
      int kp = j & 15, np = j >> 4;
      int kk = kp << 1, nn = np << 1;
      unsigned u0 = *(const unsigned*)(W + (size_t)(k0 + kk) * N + bn + nn);
      unsigned u1 = *(const unsigned*)(W + (size_t)(k0 + kk + 1) * N + bn + nn);
      unsigned t0 = (u0 & 0xffffu) | (u1 << 16);
      unsigned t1 = (u0 >> 16) | (u1 & 0xffff0000u);
      ((unsigned*)BsT)[nn * 16 + kp] = t0;
      ((unsigned*)BsT)[(nn + 1) * 16 + kp] = t1;
    }
    if (k0 + 32 < K) {   // prefetch next K-tile (global_prefetch_b8)
      __builtin_prefetch(A + (size_t)(bm + (tid >> 1)) * K + k0 + 32, 0, 0);
      __builtin_prefetch(W + (size_t)(k0 + 32 + (tid & 31)) * N + bn, 0, 0);
    }
    __syncthreads();

    FragU af[2], bfr[4];
#pragma unroll
    for (int rt = 0; rt < 2; ++rt) {
      int m = wm + (rt << 4) + ln16;
      af[rt].q[0] = *(const u32x4*)&As[(m << 5) + (hlf << 3)];
      af[rt].q[1] = *(const u32x4*)&As[(m << 5) + 16 + (hlf << 3)];
    }
#pragma unroll
    for (int ct = 0; ct < 4; ++ct) {
      int n = wn + (ct << 4) + ln16;
      bfr[ct].q[0] = *(const u32x4*)&BsT[(n << 5) + (hlf << 3)];
      bfr[ct].q[1] = *(const u32x4*)&BsT[(n << 5) + 16 + (hlf << 3)];
    }
#pragma unroll
    for (int rt = 0; rt < 2; ++rt)
#pragma unroll
      for (int ct = 0; ct < 4; ++ct)
        acc[rt][ct] = wmma_bf16(af[rt].v, bfr[ct].v, acc[rt][ct]);
    __syncthreads();
  }

#pragma unroll
  for (int rt = 0; rt < 2; ++rt)
#pragma unroll
    for (int ct = 0; ct < 4; ++ct)
#pragma unroll
      for (int v = 0; v < 8; ++v) {
        int gm = bm + wm + (rt << 4) + v + (hlf << 3);
        int gn = bn + wn + (ct << 4) + ln16;
        float val = acc[rt][ct][v] + bias[gn];
        if (FUSE == 1) val = 0.5f * val * (1.0f + erff(val * 0.70710678118f));
        if (FUSE == 2) val += resid[(size_t)gm * N + gn];
        if (outb) outb[(size_t)gm * N + gn] = f2bf(val);
        if (outf) outf[(size_t)gm * N + gn] = val;
      }
}

// ---------------------------------------------------------------- attention
// One block per (b,h); 4 waves; each wave owns 16-row query tiles, online
// softmax over 32-key chunks; WMMA for QK^T and PV.
__global__ __launch_bounds__(128) void attention_kernel(
    const unsigned short* __restrict__ qg, const unsigned short* __restrict__ kg,
    const unsigned short* __restrict__ vg, const int* __restrict__ seq,
    unsigned short* __restrict__ ctxg) {
  __shared__ alignas(16) unsigned short Ks[SP_ * DH_];   // [s][d] 28 KB
  __shared__ alignas(16) unsigned short VsT[DH_ * SP_];  // [d][s] 28 KB
  __shared__ alignas(16) unsigned short Pb[4][16 * 32];  // per-wave P buffer
  __shared__ float biasK[SP_];

  const int bh = blockIdx.x;
  const int b = bh / H_;
  const int h = bh % H_;
  const int tid = threadIdx.x;
  const int lane = tid & 31;
  const int wv = tid >> 5;
  const int hlf = lane >> 4;
  const int ln16 = lane & 15;
  const size_t base = (size_t)b * S_ * D_ + (size_t)h * DH_;
  const float scale = 0.125f;  // 1/sqrt(64)

  // K row-major (b128 copies, zero padded)
#pragma unroll
  for (int it = 0; it < 14; ++it) {
    int j = tid + (it << 7);              // 1792 u32x4 items
    int s = j >> 3, part = j & 7;
    u32x4 val = {0u, 0u, 0u, 0u};
    if (s < S_) val = *(const u32x4*)(kg + base + (size_t)s * D_ + (part << 3));
    ((u32x4*)Ks)[j] = val;
  }
  // V transposed [d][s] via register 2x2 transpose
#pragma unroll
  for (int it = 0; it < 28; ++it) {
    int j = tid + (it << 7);              // 3584 items: 32 d-pairs x 112 s-pairs
    int dp = j & 31, sp = j >> 5;
    int d = dp << 1, s = sp << 1;
    unsigned u0 = (s < S_) ? *(const unsigned*)(vg + base + (size_t)s * D_ + d) : 0u;
    unsigned u1 = (s + 1 < S_) ? *(const unsigned*)(vg + base + (size_t)(s + 1) * D_ + d) : 0u;
    unsigned t0 = (u0 & 0xffffu) | (u1 << 16);
    unsigned t1 = (u0 >> 16) | (u1 & 0xffff0000u);
    ((unsigned*)VsT)[d * (SP_ / 2) + sp] = t0;
    ((unsigned*)VsT)[(d + 1) * (SP_ / 2) + sp] = t1;
  }
  for (int i = tid; i < SP_; i += 128)
    biasK[i] = (i < S_) ? ((seq[b * S_ + i] != 0) ? 0.f : -1.0e4f) : -3.0e30f;
  __syncthreads();

  unsigned short* Pw = &Pb[wv][0];

  for (int qt = wv; qt < (SP_ / 16); qt += 4) {
    const int qbase = qt << 4;
    // Q fragments for d-chunks 0..31 and 32..63 (two b128 runs each)
    FragU qf[2];
    {
      int m = qbase + ln16;
      if (m < S_) {
        const unsigned short* qp = qg + base + (size_t)m * D_;
        qf[0].q[0] = *(const u32x4*)(qp + (hlf << 3));
        qf[0].q[1] = *(const u32x4*)(qp + 16 + (hlf << 3));
        qf[1].q[0] = *(const u32x4*)(qp + 32 + (hlf << 3));
        qf[1].q[1] = *(const u32x4*)(qp + 48 + (hlf << 3));
      } else {
        u32x4 z = {0u, 0u, 0u, 0u};
        qf[0].q[0] = z; qf[0].q[1] = z; qf[1].q[0] = z; qf[1].q[1] = z;
      }
    }
    float mrow[8], lrow[8];
    v8f acc[4];
#pragma unroll
    for (int v = 0; v < 8; ++v) { mrow[v] = -3.0e30f; lrow[v] = 0.f; }
#pragma unroll
    for (int t = 0; t < 4; ++t)
#pragma unroll
      for (int v = 0; v < 8; ++v) acc[t][v] = 0.f;

    for (int kc = 0; kc < SP_; kc += 32) {
      v8f st[2];
#pragma unroll
      for (int sub = 0; sub < 2; ++sub) {
        const int kt = kc + (sub << 4);
        v8f sc;
#pragma unroll
        for (int v = 0; v < 8; ++v) sc[v] = 0.f;
#pragma unroll
        for (int g = 0; g < 2; ++g) {          // two 32-wide d-chunks
          FragU kf;                             // B-frag of K^T (table order)
          kf.q[0] = *(const u32x4*)&Ks[(kt + ln16) * DH_ + (g << 5) + (hlf << 3)];
          kf.q[1] = *(const u32x4*)&Ks[(kt + ln16) * DH_ + (g << 5) + 16 + (hlf << 3)];
          sc = wmma_bf16(qf[g].v, kf.v, sc);
        }
        // branch-free mask: reference adds -1e4 when (k>q) or pad; -3e30 beyond S
        const int ki = kt + ln16;
        const float bK = biasK[ki];
        const float bC = fminf(bK, -1.0e4f);
#pragma unroll
        for (int v = 0; v < 8; ++v) {
          int qi = qbase + v + (hlf << 3);
          float bb = (ki > qi) ? bC : bK;
          sc[v] = sc[v] * scale + bb;
        }
        st[sub] = sc;
      }
      // online softmax update
      float mnew[8], fac[8];
#pragma unroll
      for (int v = 0; v < 8; ++v) {
        float tm = fmaxf(st[0][v], st[1][v]);
        tm = grp16_max(tm);
        mnew[v] = fmaxf(mrow[v], tm);
        fac[v] = __expf(mrow[v] - mnew[v]);
        lrow[v] *= fac[v];
      }
#pragma unroll
      for (int t = 0; t < 4; ++t)
#pragma unroll
        for (int v = 0; v < 8; ++v) acc[t][v] *= fac[v];
#pragma unroll
      for (int sub = 0; sub < 2; ++sub)
#pragma unroll
        for (int v = 0; v < 8; ++v) {
          float p = __expf(st[sub][v] - mnew[v]);
          lrow[v] += grp16_sum(p);
          Pw[((v + (hlf << 3)) << 5) + (sub << 4) + ln16] = f2bf(p);
        }
      // P (16x32) as A-frag (wave-private LDS; in-order LDS pipe)
      FragU pf;
      pf.q[0] = *(const u32x4*)&Pw[(ln16 << 5) + (hlf << 3)];
      pf.q[1] = *(const u32x4*)&Pw[(ln16 << 5) + 16 + (hlf << 3)];
#pragma unroll
      for (int t = 0; t < 4; ++t) {
        FragU vf;   // B-frag of V from transposed LDS (table order)
        const int col = (t << 4) + ln16;
        vf.q[0] = *(const u32x4*)&VsT[(size_t)col * SP_ + kc + (hlf << 3)];
        vf.q[1] = *(const u32x4*)&VsT[(size_t)col * SP_ + kc + 16 + (hlf << 3)];
        acc[t] = wmma_bf16(pf.v, vf.v, acc[t]);
      }
#pragma unroll
      for (int v = 0; v < 8; ++v) mrow[v] = mnew[v];
    }
    // normalize and store context (bf16)
#pragma unroll
    for (int t = 0; t < 4; ++t)
#pragma unroll
      for (int v = 0; v < 8; ++v) {
        int qi = qbase + v + (hlf << 3);
        if (qi < S_) {
          float den = lrow[v];
          float val = (den > 0.f) ? acc[t][v] / den : 0.f;
          ctxg[base + (size_t)qi * D_ + (t << 4) + ln16] = f2bf(val);
        }
      }
  }
}

// ---------------------------------------------------------------- gather
__global__ void gather_kernel(const float* __restrict__ x,
                              const int* __restrict__ len,
                              float* __restrict__ out) {
  int b = blockIdx.x;
  int t = len[b] - 1;
  for (int d = threadIdx.x; d < D_; d += blockDim.x)
    out[(size_t)b * D_ + d] = x[((size_t)b * S_ + t) * D_ + d];
}

// ---------------------------------------------------------------- launch
extern "C" void kernel_launch(void* const* d_in, const int* in_sizes, int n_in,
                              void* d_out, int out_size, void* d_ws,
                              size_t ws_size, hipStream_t stream) {
  (void)in_sizes; (void)n_in; (void)out_size; (void)ws_size;
  const float* item_emb = (const float*)d_in[0];
  const float* pos_emb  = (const float*)d_in[1];
  const float* ln0_g = (const float*)d_in[2];
  const float* ln0_b = (const float*)d_in[3];
  const float* Wq = (const float*)d_in[4];
  const float* bq = (const float*)d_in[5];
  const float* Wk = (const float*)d_in[6];
  const float* bk = (const float*)d_in[7];
  const float* Wv = (const float*)d_in[8];
  const float* bv = (const float*)d_in[9];
  const float* Wo = (const float*)d_in[10];
  const float* bo = (const float*)d_in[11];
  const float* ln1_g = (const float*)d_in[12];
  const float* ln1_b = (const float*)d_in[13];
  const float* W1 = (const float*)d_in[14];
  const float* b1 = (const float*)d_in[15];
  const float* W2 = (const float*)d_in[16];
  const float* b2 = (const float*)d_in[17];
  const float* ln2_g = (const float*)d_in[18];
  const float* ln2_b = (const float*)d_in[19];
  const int* item_seq = (const int*)d_in[20];
  const int* item_seq_len = (const int*)d_in[21];
  float* out = (float*)d_out;

  char* ws = (char*)d_ws;
  size_t off = 0;
  auto take = [&](size_t bytes) -> char* {
    char* p = ws + off;
    off += (bytes + 255) & ~(size_t)255;
    return p;
  };
  float* x   = (float*)take((size_t)BS_ * D_ * 4);
  float* tmp = (float*)take((size_t)BS_ * D_ * 4);
  unsigned short* xb   = (unsigned short*)take((size_t)BS_ * D_ * 2);
  unsigned short* qbuf = (unsigned short*)take((size_t)BS_ * D_ * 2);
  unsigned short* kbuf = (unsigned short*)take((size_t)BS_ * D_ * 2);
  unsigned short* vbuf = (unsigned short*)take((size_t)BS_ * D_ * 2);
  unsigned short* cbuf = (unsigned short*)take((size_t)BS_ * D_ * 2);
  unsigned short* ff1  = (unsigned short*)take((size_t)BS_ * F_ * 2);
  unsigned short* wq_bf = (unsigned short*)take((size_t)L_ * D_ * D_ * 2);
  unsigned short* wk_bf = (unsigned short*)take((size_t)L_ * D_ * D_ * 2);
  unsigned short* wv_bf = (unsigned short*)take((size_t)L_ * D_ * D_ * 2);
  unsigned short* wo_bf = (unsigned short*)take((size_t)L_ * D_ * D_ * 2);
  unsigned short* w1_bf = (unsigned short*)take((size_t)L_ * D_ * F_ * 2);
  unsigned short* w2_bf = (unsigned short*)take((size_t)L_ * F_ * D_ * 2);

  // weights -> bf16 (cheap vs GEMMs; deterministic each call)
  f32_to_bf16_kernel<<<dim3(1024), dim3(256), 0, stream>>>(Wq, wq_bf, (size_t)L_ * D_ * D_);
  f32_to_bf16_kernel<<<dim3(1024), dim3(256), 0, stream>>>(Wk, wk_bf, (size_t)L_ * D_ * D_);
  f32_to_bf16_kernel<<<dim3(1024), dim3(256), 0, stream>>>(Wv, wv_bf, (size_t)L_ * D_ * D_);
  f32_to_bf16_kernel<<<dim3(1024), dim3(256), 0, stream>>>(Wo, wo_bf, (size_t)L_ * D_ * D_);
  f32_to_bf16_kernel<<<dim3(2048), dim3(256), 0, stream>>>(W1, w1_bf, (size_t)L_ * D_ * F_);
  f32_to_bf16_kernel<<<dim3(2048), dim3(256), 0, stream>>>(W2, w2_bf, (size_t)L_ * F_ * D_);

  embed_ln_kernel<<<dim3(BS_ / 8), dim3(256), 0, stream>>>(
      item_emb, pos_emb, ln0_g, ln0_b, item_seq, x, xb);

  dim3 g256(D_ / 128, BS_ / 64);   // (2, 1600)
  dim3 g1024(F_ / 128, BS_ / 64);  // (8, 1600)

  for (int l = 0; l < L_; ++l) {
    const size_t wDD = (size_t)l * D_ * D_;
    const size_t wDF = (size_t)l * D_ * F_;
    gemm_bf16_kernel<0><<<g256, 128, 0, stream>>>(
        xb, wq_bf + wDD, bq + l * D_, nullptr, qbuf, nullptr, BS_, D_, D_);
    gemm_bf16_kernel<0><<<g256, 128, 0, stream>>>(
        xb, wk_bf + wDD, bk + l * D_, nullptr, kbuf, nullptr, BS_, D_, D_);
    gemm_bf16_kernel<0><<<g256, 128, 0, stream>>>(
        xb, wv_bf + wDD, bv + l * D_, nullptr, vbuf, nullptr, BS_, D_, D_);
    attention_kernel<<<dim3(B_ * H_), dim3(128), 0, stream>>>(
        qbuf, kbuf, vbuf, item_seq, cbuf);
    gemm_bf16_kernel<2><<<g256, 128, 0, stream>>>(
        cbuf, wo_bf + wDD, bo + l * D_, x, nullptr, tmp, BS_, D_, D_);
    ln_kernel<<<dim3(BS_ / 8), dim3(256), 0, stream>>>(
        tmp, ln1_g + l * D_, ln1_b + l * D_, x, xb);
    gemm_bf16_kernel<1><<<g1024, 128, 0, stream>>>(
        xb, w1_bf + wDF, b1 + l * F_, nullptr, ff1, nullptr, BS_, F_, D_);
    gemm_bf16_kernel<2><<<g256, 128, 0, stream>>>(
        ff1, w2_bf + wDF, b2 + l * D_, x, nullptr, tmp, BS_, D_, F_);
    ln_kernel<<<dim3(BS_ / 8), dim3(256), 0, stream>>>(
        tmp, ln2_g + l * D_, ln2_b + l * D_, x, xb);
  }
  gather_kernel<<<dim3(B_), dim3(256), 0, stream>>>(x, item_seq_len, out);
}
